// NearestClassMean_34213709479984
// MI455X (gfx1250) — matrix operand order repaired and checked by reading, not compiled, not run on hardware
//
#include <hip/hip_runtime.h>

// MI455X (gfx1250) nearest-class-mean scores.
// scores = 2*X.muK^T - |x|^2 - |mu|^2 via v_wmma_f32_16x16x32_f16,
// single kernel: block owns 64 rows, loops all N -> row-min/mask stay local.
// B-tile global loads are software-pipelined (register prefetch) so L2 latency
// hides under the WMMA phase instead of stalling the staging barrier.

typedef __attribute__((ext_vector_type(16))) _Float16 v16h;
typedef __attribute__((ext_vector_type(8)))  float    v8f;

#define M_TOT   16384
#define N_TOT   1000
#define D_TOT   512
#define MTILE   64
#define NSTRIP  256
#define KCH     32
#define NCHUNKS (D_TOT / KCH)     // 16
#define A_STRIDE 520              // fp16 elems per row: 512 + 8 pad (16B-aligned runs, 260 dwords)
#define B_STRIDE 40               // fp16 elems per row: 32 + 8 pad (20 dwords)

union FragH { v16h v; uint4 q[2]; };
union Pack8 { _Float16 h[8]; uint4 q; };

__device__ __forceinline__ void load_b_regs(const float* __restrict__ muK,
                                            int nbase, int cb, int qd, int kc,
                                            float4 pf0[4], float4 pf1[4]) {
    #pragma unroll
    for (int cc = 0; cc < 4; ++cc) {
        const int gcol = nbase + cb + cc * 64;
        float4 a = {0.f, 0.f, 0.f, 0.f}, b = {0.f, 0.f, 0.f, 0.f};
        if (gcol < N_TOT) {
            const float* Br = muK + (size_t)gcol * D_TOT + kc + qd * 8;
            a = *(const float4*)(Br);
            b = *(const float4*)(Br + 4);
        }
        pf0[cc] = a;
        pf1[cc] = b;
    }
}

__global__ __launch_bounds__(256)
void ncm_wmma_kernel(const float* __restrict__ X,
                     const float* __restrict__ muK,
                     const int*   __restrict__ cK,
                     float*       __restrict__ out) {
    __shared__ __align__(16) _Float16 Alds[MTILE * A_STRIDE];    // 65 KB, all K for 64 rows
    __shared__ __align__(16) _Float16 Blds[NSTRIP * B_STRIDE];   // 20 KB, one 32-K chunk for 256 cols
    __shared__ float xsq_lds[MTILE];
    __shared__ float musq_lds[NSTRIP];
    __shared__ float parts[4][MTILE];
    __shared__ float rowmin_lds[MTILE];

    const int tid  = threadIdx.x;
    const int m0   = blockIdx.x * MTILE;
    const int lane = tid & 31;
    const int w    = tid >> 5;
    const int wm   = w >> 2;            // 0..1 : 32-row group
    const int wn   = w & 3;             // 0..3 : 64-col group within strip
    const int lrow = lane & 15;
    const int lhalf = lane >> 4;        // 0/1

    // ---------- Stage A: X tile (64 x 512) fp32 -> fp16 in LDS, plus |x|^2 ----------
    {
        const int r  = tid >> 2;        // 0..63 row
        const int qd = tid & 3;         // k-quarter within each 32-chunk
        const float* Xr = X + (size_t)(m0 + r) * D_TOT + qd * 8;
        float ss = 0.f;
        #pragma unroll 4
        for (int c = 0; c < NCHUNKS; ++c) {
            float4 f0 = *(const float4*)(Xr + c * KCH);
            float4 f1 = *(const float4*)(Xr + c * KCH + 4);
            ss = fmaf(f0.x, f0.x, ss); ss = fmaf(f0.y, f0.y, ss);
            ss = fmaf(f0.z, f0.z, ss); ss = fmaf(f0.w, f0.w, ss);
            ss = fmaf(f1.x, f1.x, ss); ss = fmaf(f1.y, f1.y, ss);
            ss = fmaf(f1.z, f1.z, ss); ss = fmaf(f1.w, f1.w, ss);
            Pack8 u;
            u.h[0] = (_Float16)f0.x; u.h[1] = (_Float16)f0.y;
            u.h[2] = (_Float16)f0.z; u.h[3] = (_Float16)f0.w;
            u.h[4] = (_Float16)f1.x; u.h[5] = (_Float16)f1.y;
            u.h[6] = (_Float16)f1.z; u.h[7] = (_Float16)f1.w;
            *(uint4*)(&Alds[r * A_STRIDE + c * KCH + qd * 8]) = u.q;
        }
        ss += __shfl_xor(ss, 1);
        ss += __shfl_xor(ss, 2);
        if (qd == 0) xsq_lds[r] = ss;
    }
    __syncthreads();

    // Per-lane |x|^2 for the rows this lane finalizes (VGPR i -> row base+lhalf*8+i)
    float xr[2][8];
    #pragma unroll
    for (int tm = 0; tm < 2; ++tm)
        #pragma unroll
        for (int i = 0; i < 8; ++i)
            xr[tm][i] = xsq_lds[wm * 32 + tm * 16 + lhalf * 8 + i];

    float rmin[2][8];
    #pragma unroll
    for (int tm = 0; tm < 2; ++tm)
        #pragma unroll
        for (int i = 0; i < 8; ++i)
            rmin[tm][i] = INFINITY;

    const v8f vzero = {};
    const int cb = tid >> 2;            // 0..63 column-in-strip base
    const int qd = tid & 3;             // k-quarter

    // ---------- Loop over 4 column strips of 256 ----------
    for (int ns = 0; ns < 4; ++ns) {
        const int nbase = ns * NSTRIP;
        v8f acc[2][4];
        #pragma unroll
        for (int tm = 0; tm < 2; ++tm)
            #pragma unroll
            for (int tn = 0; tn < 4; ++tn)
                acc[tm][tn] = vzero;

        float mq_acc[4] = {0.f, 0.f, 0.f, 0.f};

        // Prefetch chunk 0 of this strip into registers
        float4 pf0[4], pf1[4];
        load_b_regs(muK, nbase, cb, qd, 0, pf0, pf1);

        for (int ch = 0; ch < NCHUNKS; ++ch) {
            const int kc = ch * KCH;
            __syncthreads();            // previous compute done -> Blds reusable
            // Convert prefetched B regs -> fp16 in LDS, accumulate |mu|^2
            #pragma unroll
            for (int cc = 0; cc < 4; ++cc) {
                const float4 f0 = pf0[cc];
                const float4 f1 = pf1[cc];
                float s = mq_acc[cc];
                s = fmaf(f0.x, f0.x, s); s = fmaf(f0.y, f0.y, s);
                s = fmaf(f0.z, f0.z, s); s = fmaf(f0.w, f0.w, s);
                s = fmaf(f1.x, f1.x, s); s = fmaf(f1.y, f1.y, s);
                s = fmaf(f1.z, f1.z, s); s = fmaf(f1.w, f1.w, s);
                mq_acc[cc] = s;
                Pack8 u;
                u.h[0] = (_Float16)f0.x; u.h[1] = (_Float16)f0.y;
                u.h[2] = (_Float16)f0.z; u.h[3] = (_Float16)f0.w;
                u.h[4] = (_Float16)f1.x; u.h[5] = (_Float16)f1.y;
                u.h[6] = (_Float16)f1.z; u.h[7] = (_Float16)f1.w;
                *(uint4*)(&Blds[(cb + cc * 64) * B_STRIDE + qd * 8]) = u.q;
            }
            // Issue next chunk's global loads now; latency hides under WMMA phase
            if (ch + 1 < NCHUNKS)
                load_b_regs(muK, nbase, cb, qd, kc + KCH, pf0, pf1);
            __syncthreads();            // staged chunk visible

            // A frags: per-lane runs [kc+lhalf*8, +8) and [kc+16+lhalf*8, +8)  (ISA 16-bit A 16x32 layout)
            FragH af[2];
            #pragma unroll
            for (int tm = 0; tm < 2; ++tm) {
                const _Float16* p = &Alds[(wm * 32 + tm * 16 + lrow) * A_STRIDE + kc + lhalf * 8];
                af[tm].q[0] = *(const uint4*)(p);
                af[tm].q[1] = *(const uint4*)(p + 16);
            }
            // B frags: per-lane runs [lhalf*8, +8) and [16+lhalf*8, +8) (same k order as A), then 8 WMMAs
            #pragma unroll
            for (int tn = 0; tn < 4; ++tn) {
                FragH bf;
                const _Float16* p = &Blds[(wn * 64 + tn * 16 + lrow) * B_STRIDE + lhalf * 8];
                bf.q[0] = *(const uint4*)(p);
                bf.q[1] = *(const uint4*)(p + 16);
                #pragma unroll
                for (int tm = 0; tm < 2; ++tm)
                    acc[tm][tn] = __builtin_amdgcn_wmma_f32_16x16x32_f16(
                        false, af[tm].v, false, bf.v, (short)0, acc[tm][tn], false, false);
            }
        }

        // |mu|^2 for this strip: quad reduce, publish
        #pragma unroll
        for (int cc = 0; cc < 4; ++cc) {
            float s = mq_acc[cc];
            s += __shfl_xor(s, 1);
            s += __shfl_xor(s, 2);
            if (qd == 0) musq_lds[cb + cc * 64] = s;
        }
        __syncthreads();

        // Finalize: scores = 2*acc - |x|^2 - |mu|^2 ; store + running row min (valid cols only)
        #pragma unroll
        for (int tn = 0; tn < 4; ++tn) {
            const int col = nbase + wn * 64 + tn * 16 + lrow;
            const float mq = musq_lds[wn * 64 + tn * 16 + lrow];
            if (col < N_TOT) {
                #pragma unroll
                for (int tm = 0; tm < 2; ++tm) {
                    const int rbase = m0 + wm * 32 + tm * 16 + lhalf * 8;
                    #pragma unroll
                    for (int i = 0; i < 8; ++i) {
                        float s = 2.0f * acc[tm][tn][i] - xr[tm][i] - mq;
                        out[(size_t)(rbase + i) * N_TOT + col] = s;
                        rmin[tm][i] = fminf(rmin[tm][i], s);
                    }
                }
            }
        }
    }

    // Order pass-1 stores before masked overwrites
    __threadfence();

    // Row-min: reduce across the 16 cols held per lane-half, combine across wn waves
    #pragma unroll
    for (int tm = 0; tm < 2; ++tm)
        #pragma unroll
        for (int i = 0; i < 8; ++i) {
            float v = rmin[tm][i];
            v = fminf(v, __shfl_xor(v, 1, 16));
            v = fminf(v, __shfl_xor(v, 2, 16));
            v = fminf(v, __shfl_xor(v, 4, 16));
            v = fminf(v, __shfl_xor(v, 8, 16));
            if (lrow == 0) parts[wn][wm * 32 + tm * 16 + lhalf * 8 + i] = v;
        }
    __syncthreads();
    if (tid < MTILE) {
        float v = fminf(fminf(parts[0][tid], parts[1][tid]),
                        fminf(parts[2][tid], parts[3][tid]));
        rowmin_lds[tid] = v - 1.0f;
    }
    __syncthreads();

    // Masked columns (cK==0) -> rowmin - 1
    for (int idx = tid; idx < MTILE * N_TOT; idx += 256) {
        const int n = idx >> 6;       // 0..999
        const int r = idx & 63;
        if (cK[n] == 0)
            out[(size_t)(m0 + r) * N_TOT + n] = rowmin_lds[r];
    }
}

extern "C" void kernel_launch(void* const* d_in, const int* in_sizes, int n_in,
                              void* d_out, int out_size, void* d_ws, size_t ws_size,
                              hipStream_t stream) {
    (void)in_sizes; (void)n_in; (void)d_ws; (void)ws_size; (void)out_size;
    const float* X   = (const float*)d_in[0];
    const float* muK = (const float*)d_in[1];
    const int*   cK  = (const int*)d_in[2];
    float* out = (float*)d_out;
    dim3 grid(M_TOT / MTILE);   // 256 blocks
    dim3 block(256);            // 8 waves
    ncm_wmma_kernel<<<grid, block, 0, stream>>>(X, muK, cK, out);
}
